// SwinTransformerBlock_27822798143628
// MI455X (gfx1250) — compile-verified, matrix operand order checked by reading, not run on hardware
//
#include <hip/hip_runtime.h>
#include <hip/hip_bf16.h>

// ---------------------------------------------------------------------------
// Swin Transformer block for MI455X (gfx1250), wave32 + v_wmma_f32_16x16x32_f16
// + GLOBAL_LOAD_ASYNC_TO_LDS_B128 for bulk tile staging (ASYNCcnt path)
// ---------------------------------------------------------------------------

#define DIM       256
#define HEADS     8
#define HEAD_DIM  32
#define WS        8
#define NWIN_TOK  64          // N = WS*WS
#define SHIFT     4
#define MLP_HID   1024
#define HH        128
#define WW        128
#define NWIN      4096        // 16 * (128/8) * (128/8)
#define NTOK      (16 * 16384)
#define SCALE_F   0.17677669529663687f   // 32^-0.5
#define VT_LD     80          // padded row stride (halves) for transposed V

typedef _Float16 half_t;
typedef __attribute__((ext_vector_type(16))) _Float16 v16h;
typedef __attribute__((ext_vector_type(8)))  _Float16 v8h;
typedef __attribute__((ext_vector_type(8)))  float    v8f;

union Frag16 { v16h v; v8h h[2]; };

// A/B fragment, row-contiguous source (A: M x K row-major; B: weight stored
// N-major as Wt[N][K]).  lane&15 -> row, hi lane group -> K interleave:
//  halves 0..7  = K (hi*8)+0..7 ;  halves 8..15 = K 16+(hi*8)+0..7
__device__ __forceinline__ v16h load_frag_rm(const half_t* __restrict__ base,
                                             int ld, int lane) {
    int r = lane & 15, hi = lane >> 4;
    const half_t* p = base + r * ld + hi * 8;
    Frag16 f;
    f.h[0] = *(const v8h*)(p);
    f.h[1] = *(const v8h*)(p + 16);
    return f.v;
}

__device__ __forceinline__ v8f wmma16(v16h a, v16h b, v8f c) {
    return __builtin_amdgcn_wmma_f32_16x16x32_f16(
        false, a, false, b, (short)0, c, false, false);
}

// Async global -> LDS copy, 16 bytes per lane.  VGPR %0 holds the per-lane LDS
// byte offset (relative to the workgroup LDS base), %1 the 64-bit global addr.
__device__ __forceinline__ void g2lds_b128(unsigned lds_off, const void* gsrc) {
    asm volatile("global_load_async_to_lds_b128 %0, %1, off"
                 :: "v"(lds_off), "v"(gsrc) : "memory");
}
__device__ __forceinline__ void wait_async0() {
    asm volatile("s_wait_asynccnt 0" ::: "memory");
}

// ---------------------------------------------------------------------------
// Prep: fp32 [K][N] weight -> f16 transposed [N][K]
// ---------------------------------------------------------------------------
__global__ void k_transpose_w(const float* __restrict__ in,
                              half_t* __restrict__ outw, int K, int Nn) {
    int idx = blockIdx.x * 256 + threadIdx.x;
    if (idx >= K * Nn) return;
    int n = idx / K, k = idx % K;
    outw[idx] = (half_t)in[k * Nn + n];
}

// Prep: bias[h][n][m] = rpb_table[relidx(n,m)][h]
__global__ void k_build_bias(const float* __restrict__ rpb,
                             float* __restrict__ bias) {
    int idx = blockIdx.x * 256 + threadIdx.x;
    if (idx >= HEADS * NWIN_TOK * NWIN_TOK) return;
    int h = idx >> 12, nm = idx & 4095, n = nm >> 6, m = nm & 63;
    int yn = n >> 3, xn = n & 7, ym = m >> 3, xm = m & 7;
    int rel = (yn - ym + WS - 1) * (2 * WS - 1) + (xn - xm + WS - 1);
    bias[idx] = rpb[rel * HEADS + h];
}

// ---------------------------------------------------------------------------
// LN1 + cyclic shift + window partition -> f16 windows [NWIN][64][256]
// one block per window, 4 lanes per token
// ---------------------------------------------------------------------------
__global__ void k_ln1_window(const float* __restrict__ x,
                             const float* __restrict__ g,
                             const float* __restrict__ bb,
                             half_t* __restrict__ xw) {
    int w = blockIdx.x;
    int b = w >> 8, wi = w & 255, wr = wi >> 4, wc = wi & 15;
    int n = threadIdx.x >> 2;   // token in window
    int q = threadIdx.x & 3;    // quarter of the channel dim
    int hs = wr * WS + (n >> 3), ws2 = wc * WS + (n & 7);
    int hsrc = (hs + SHIFT) & (HH - 1), wsrc = (ws2 + SHIFT) & (WW - 1);
    const float* row = x + (((size_t)b * HH + hsrc) * WW + wsrc) * DIM;

    float4 v[16];
    float s = 0.f, ss = 0.f;
#pragma unroll
    for (int i = 0; i < 16; ++i) {
        v[i] = ((const float4*)row)[q * 16 + i];
        s  += v[i].x + v[i].y + v[i].z + v[i].w;
        ss += v[i].x * v[i].x + v[i].y * v[i].y + v[i].z * v[i].z + v[i].w * v[i].w;
    }
    s  += __shfl_xor(s, 1, 32);  s  += __shfl_xor(s, 2, 32);
    ss += __shfl_xor(ss, 1, 32); ss += __shfl_xor(ss, 2, 32);
    float mu  = s * (1.0f / DIM);
    float var = ss * (1.0f / DIM) - mu * mu;
    float inv = rsqrtf(var + 1e-5f);

    half_t* dst = xw + ((size_t)w * NWIN_TOK + n) * DIM + q * 64;
#pragma unroll
    for (int i = 0; i < 16; ++i) {
        int c0 = q * 64 + i * 4;
        dst[i * 4 + 0] = (half_t)((v[i].x - mu) * inv * g[c0 + 0] + bb[c0 + 0]);
        dst[i * 4 + 1] = (half_t)((v[i].y - mu) * inv * g[c0 + 1] + bb[c0 + 1]);
        dst[i * 4 + 2] = (half_t)((v[i].z - mu) * inv * g[c0 + 2] + bb[c0 + 2]);
        dst[i * 4 + 3] = (half_t)((v[i].w - mu) * inv * g[c0 + 3] + bb[c0 + 3]);
    }
}

// ---------------------------------------------------------------------------
// Windowed attention: one block (8 waves) per window.
// LDS: Q (64x256), K (64x256), Vt (256xVT_LD padded), X/O (64x256) = 136 KB.
// P (f16, per-head 64x64) overlays the dead Q+K region.
// ---------------------------------------------------------------------------
#define LDS_Q_OFF   0
#define LDS_K_OFF   (NWIN_TOK * DIM)                 // halves
#define LDS_VT_OFF  (2 * NWIN_TOK * DIM)
#define LDS_X_OFF   (2 * NWIN_TOK * DIM + DIM * VT_LD)
#define LDS_ATTN_HALVES (LDS_X_OFF + NWIN_TOK * DIM)

__global__ void k_window_attn(const half_t* __restrict__ xw,
                              const half_t* __restrict__ qkvWt,
                              const float*  __restrict__ qkv_b,
                              const half_t* __restrict__ projWt,
                              const float*  __restrict__ proj_b,
                              const float*  __restrict__ biasMat,
                              float* __restrict__ out_attn) {
    extern __shared__ char smem[];
    half_t* ldsQ  = (half_t*)smem + LDS_Q_OFF;
    half_t* ldsK  = (half_t*)smem + LDS_K_OFF;
    half_t* ldsVt = (half_t*)smem + LDS_VT_OFF;      // [256][VT_LD] transposed V
    half_t* ldsX  = (half_t*)smem + LDS_X_OFF;       // X, reused for O
    half_t* ldsP  = ldsQ;                            // head h -> +h*64*64

    int w    = blockIdx.x;
    int lane = threadIdx.x & 31;
    int wid  = threadIdx.x >> 5;

    // ---- async load of X window (32 KB) straight into LDS ----
    {
        const char* src = (const char*)(xw + (size_t)w * NWIN_TOK * DIM);
        unsigned dstOff = (unsigned)(LDS_X_OFF * sizeof(half_t));
#pragma unroll
        for (int i = 0; i < (NWIN_TOK * DIM * 2) / (256 * 16); ++i) {
            int e = i * 256 + threadIdx.x;
            g2lds_b128(dstOff + e * 16, src + e * 16);
        }
        // warm L2 with the weight panels while the async copy runs
        __builtin_prefetch(qkvWt + (size_t)wid * 12288, 0, 1);
        __builtin_prefetch(projWt + (size_t)wid * 8192, 0, 1);
        wait_async0();
    }
    __syncthreads();

    // ---- QKV GEMM: 64x768, 4x48 tiles ----
    for (int t = wid; t < 4 * 48; t += 8) {
        int mt = t / 48, nt = t % 48;
        v8f acc = {};
#pragma unroll
        for (int kk = 0; kk < 8; ++kk) {
            v16h a = load_frag_rm(ldsX + mt * 16 * DIM + kk * 32, DIM, lane);
            v16h b = load_frag_rm(qkvWt + (nt * 16) * DIM + kk * 32, DIM, lane);
            acc = wmma16(a, b, acc);
        }
        int n15 = lane & 15, hi = lane >> 4;
        int ncol = nt * 16 + n15;
        float bias = qkv_b[ncol];
        int sec = ncol >> 8;                 // wave-uniform: 0=q 1=k 2=v
        int cc  = ncol & 255;
        if (sec == 2) {
            // V transposed, one packed 16B store per tile per lane
            v8h pk;
#pragma unroll
            for (int r = 0; r < 8; ++r) pk[r] = (half_t)(acc[r] + bias);
            *(v8h*)(ldsVt + cc * VT_LD + mt * 16 + hi * 8) = pk;
        } else {
            half_t* dst = (sec == 0) ? ldsQ : ldsK;
            float scl = (sec == 0) ? SCALE_F : 1.0f;
#pragma unroll
            for (int r = 0; r < 8; ++r) {
                int m = mt * 16 + hi * 8 + r;
                dst[m * DIM + cc] = (half_t)((acc[r] + bias) * scl);
            }
        }
    }
    __syncthreads();

    // ---- per-head attention: wave wid handles head h = wid ----
    int h = wid;
    v8f S[16];                               // [mt][jt] 64x64 scores in f32
    {
        const float* bp = biasMat + (size_t)h * NWIN_TOK * NWIN_TOK;
        int n15 = lane & 15, hi = lane >> 4;
#pragma unroll
        for (int mt = 0; mt < 4; ++mt) {
            v16h a = load_frag_rm(ldsQ + mt * 16 * DIM + h * HEAD_DIM, DIM, lane);
#pragma unroll
            for (int jt = 0; jt < 4; ++jt) {
                v16h b = load_frag_rm(ldsK + jt * 16 * DIM + h * HEAD_DIM, DIM, lane);
                v8f c = {};
                c = wmma16(a, b, c);
#pragma unroll
                for (int r = 0; r < 8; ++r)
                    c[r] += bp[(mt * 16 + hi * 8 + r) * NWIN_TOK + jt * 16 + n15];
                S[mt * 4 + jt] = c;
            }
        }
    }

    // ---- softmax over columns (rows live in VGPR index + lane-half) ----
#pragma unroll
    for (int mt = 0; mt < 4; ++mt) {
#pragma unroll
        for (int r = 0; r < 8; ++r) {
            float m = -3.0e30f;
#pragma unroll
            for (int jt = 0; jt < 4; ++jt) m = fmaxf(m, S[mt * 4 + jt][r]);
            m = fmaxf(m, __shfl_xor(m, 1, 32));
            m = fmaxf(m, __shfl_xor(m, 2, 32));
            m = fmaxf(m, __shfl_xor(m, 4, 32));
            m = fmaxf(m, __shfl_xor(m, 8, 32));
            float s = 0.f;
#pragma unroll
            for (int jt = 0; jt < 4; ++jt) {
                float e = expf(S[mt * 4 + jt][r] - m);
                S[mt * 4 + jt][r] = e;
                s += e;
            }
            s += __shfl_xor(s, 1, 32);
            s += __shfl_xor(s, 2, 32);
            s += __shfl_xor(s, 4, 32);
            s += __shfl_xor(s, 8, 32);
            float inv = 1.0f / s;
#pragma unroll
            for (int jt = 0; jt < 4; ++jt) S[mt * 4 + jt][r] *= inv;
        }
    }

    __syncthreads();                          // all waves done reading Q/K

    // ---- P (f16) into LDS over the dead Q/K region ----
    half_t* P = ldsP + h * NWIN_TOK * NWIN_TOK;
    {
        int n15 = lane & 15, hi = lane >> 4;
#pragma unroll
        for (int mt = 0; mt < 4; ++mt)
#pragma unroll
            for (int jt = 0; jt < 4; ++jt)
#pragma unroll
                for (int r = 0; r < 8; ++r)
                    P[(mt * 16 + hi * 8 + r) * NWIN_TOK + jt * 16 + n15] =
                        (half_t)S[mt * 4 + jt][r];
    }

    // ---- O_h = P (64x64) @ V_h (64x32), written into ldsX region ----
    {
        int n15 = lane & 15, hi = lane >> 4;
#pragma unroll
        for (int mt = 0; mt < 4; ++mt)
#pragma unroll
            for (int dt = 0; dt < 2; ++dt) {
                v8f acc = {};
#pragma unroll
                for (int ks = 0; ks < 2; ++ks) {
                    v16h a = load_frag_rm(P + mt * 16 * NWIN_TOK + ks * 32,
                                          NWIN_TOK, lane);
                    v16h b = load_frag_rm(
                        ldsVt + (h * HEAD_DIM + dt * 16) * VT_LD + ks * 32,
                        VT_LD, lane);
                    acc = wmma16(a, b, acc);
                }
#pragma unroll
                for (int r = 0; r < 8; ++r)
                    ldsX[(mt * 16 + hi * 8 + r) * DIM + h * HEAD_DIM + dt * 16 + n15] =
                        (half_t)acc[r];
            }
    }
    __syncthreads();

    // ---- proj GEMM + inverse shift scatter (fp32 out) ----
    int b = w >> 8, wi = w & 255, wr = wi >> 4, wc = wi & 15;
    for (int t = wid; t < 4 * 16; t += 8) {
        int mt = t >> 4, nt = t & 15;
        v8f acc = {};
#pragma unroll
        for (int kk = 0; kk < 8; ++kk) {
            v16h a  = load_frag_rm(ldsX + mt * 16 * DIM + kk * 32, DIM, lane);
            v16h bm = load_frag_rm(projWt + (nt * 16) * DIM + kk * 32, DIM, lane);
            acc = wmma16(a, bm, acc);
        }
        int n15 = lane & 15, hi = lane >> 4;
        int col = nt * 16 + n15;
        float pb = proj_b[col];
#pragma unroll
        for (int r = 0; r < 8; ++r) {
            int n  = mt * 16 + hi * 8 + r;
            int hs = wr * WS + (n >> 3), ws2 = wc * WS + (n & 7);
            int hd = (hs + SHIFT) & (HH - 1), wd = (ws2 + SHIFT) & (WW - 1);
            out_attn[(((size_t)b * HH + hd) * WW + wd) * DIM + col] = acc[r] + pb;
        }
    }
}

// ---------------------------------------------------------------------------
// LN2 of attention output -> f16 (reuses xw buffer)
// ---------------------------------------------------------------------------
__global__ void k_ln2(const float* __restrict__ att,
                      const float* __restrict__ g,
                      const float* __restrict__ bb,
                      half_t* __restrict__ h2) {
    size_t tok = (size_t)blockIdx.x * 64 + (threadIdx.x >> 2);
    int q = threadIdx.x & 3;
    const float* row = att + tok * DIM;
    float4 v[16];
    float s = 0.f, ss = 0.f;
#pragma unroll
    for (int i = 0; i < 16; ++i) {
        v[i] = ((const float4*)row)[q * 16 + i];
        s  += v[i].x + v[i].y + v[i].z + v[i].w;
        ss += v[i].x * v[i].x + v[i].y * v[i].y + v[i].z * v[i].z + v[i].w * v[i].w;
    }
    s  += __shfl_xor(s, 1, 32);  s  += __shfl_xor(s, 2, 32);
    ss += __shfl_xor(ss, 1, 32); ss += __shfl_xor(ss, 2, 32);
    float mu  = s * (1.0f / DIM);
    float inv = rsqrtf(ss * (1.0f / DIM) - mu * mu + 1e-5f);
    half_t* dst = h2 + tok * DIM + q * 64;
#pragma unroll
    for (int i = 0; i < 16; ++i) {
        int c0 = q * 64 + i * 4;
        dst[i * 4 + 0] = (half_t)((v[i].x - mu) * inv * g[c0 + 0] + bb[c0 + 0]);
        dst[i * 4 + 1] = (half_t)((v[i].y - mu) * inv * g[c0 + 1] + bb[c0 + 1]);
        dst[i * 4 + 2] = (half_t)((v[i].z - mu) * inv * g[c0 + 2] + bb[c0 + 2]);
        dst[i * 4 + 3] = (half_t)((v[i].w - mu) * inv * g[c0 + 3] + bb[c0 + 3]);
    }
}

// ---------------------------------------------------------------------------
// MLP (gelu exact) + final residual add: out = x + att + mlp(h2)
// one block per 64 tokens; LDS: h2 tile 32 KB + hidden 128 KB (160 KB dynamic)
// ---------------------------------------------------------------------------
__global__ void k_mlp_final(const half_t* __restrict__ h2,
                            const half_t* __restrict__ w1t,
                            const float*  __restrict__ b1,
                            const half_t* __restrict__ w2t,
                            const float*  __restrict__ b2,
                            const float*  __restrict__ x,
                            const float*  __restrict__ att,
                            float* __restrict__ out) {
    extern __shared__ char smem[];
    half_t* ldsH   = (half_t*)smem;           // 64x256 at LDS offset 0
    half_t* ldsHid = ldsH + 64 * DIM;         // 64x1024

    size_t t0 = (size_t)blockIdx.x * 64;
    int lane = threadIdx.x & 31, wid = threadIdx.x >> 5;

    // ---- async load of the h2 tile (32 KB) straight into LDS ----
    {
        const char* src = (const char*)(h2 + t0 * DIM);
#pragma unroll
        for (int i = 0; i < (64 * DIM * 2) / (256 * 16); ++i) {
            int e = i * 256 + threadIdx.x;
            g2lds_b128((unsigned)(e * 16), src + e * 16);
        }
        __builtin_prefetch(w1t + (size_t)wid * 32768, 0, 1);
        __builtin_prefetch(w2t + (size_t)wid * 32768, 0, 1);
        wait_async0();
    }
    __syncthreads();

    // GEMM1 + GELU: 64x1024, 4x64 tiles
    for (int t = wid; t < 4 * 64; t += 8) {
        int mt = t >> 6, nt = t & 63;
        v8f acc = {};
#pragma unroll
        for (int kk = 0; kk < 8; ++kk) {
            v16h a = load_frag_rm(ldsH + mt * 16 * DIM + kk * 32, DIM, lane);
            v16h b = load_frag_rm(w1t + (nt * 16) * DIM + kk * 32, DIM, lane);
            acc = wmma16(a, b, acc);
        }
        int n15 = lane & 15, hi = lane >> 4;
        float bias = b1[nt * 16 + n15];
#pragma unroll
        for (int r = 0; r < 8; ++r) {
            float v = acc[r] + bias;
            v = 0.5f * v * (1.0f + erff(v * 0.70710678118654752f));
            ldsHid[(mt * 16 + hi * 8 + r) * MLP_HID + nt * 16 + n15] = (half_t)v;
        }
    }
    __syncthreads();

    // GEMM2: 64x256, K=1024, fused final residual add
    for (int t = wid; t < 4 * 16; t += 8) {
        int mt = t >> 4, nt = t & 15;
        v8f acc = {};
#pragma unroll
        for (int kk = 0; kk < 32; ++kk) {
            v16h a = load_frag_rm(ldsHid + mt * 16 * MLP_HID + kk * 32, MLP_HID, lane);
            v16h b = load_frag_rm(w2t + (nt * 16) * MLP_HID + kk * 32, MLP_HID, lane);
            acc = wmma16(a, b, acc);
        }
        int n15 = lane & 15, hi = lane >> 4;
        int col = nt * 16 + n15;
        float bias = b2[col];
#pragma unroll
        for (int r = 0; r < 8; ++r) {
            size_t idx = (t0 + mt * 16 + hi * 8 + r) * DIM + col;
            out[idx] = x[idx] + att[idx] + acc[r] + bias;
        }
    }
}

// ---------------------------------------------------------------------------
extern "C" void kernel_launch(void* const* d_in, const int* in_sizes, int n_in,
                              void* d_out, int out_size, void* d_ws, size_t ws_size,
                              hipStream_t stream) {
    const float* x       = (const float*)d_in[0];
    const float* qkv_w   = (const float*)d_in[1];
    const float* qkv_b   = (const float*)d_in[2];
    const float* proj_w  = (const float*)d_in[3];
    const float* proj_b  = (const float*)d_in[4];
    const float* rpb     = (const float*)d_in[5];
    const float* ln1_g   = (const float*)d_in[6];
    const float* ln1_b   = (const float*)d_in[7];
    const float* ln2_g   = (const float*)d_in[8];
    const float* ln2_b   = (const float*)d_in[9];
    const float* mlp_w1  = (const float*)d_in[10];
    const float* mlp_b1  = (const float*)d_in[11];
    const float* mlp_w2  = (const float*)d_in[12];
    const float* mlp_b2  = (const float*)d_in[13];
    float* out = (float*)d_out;

    // ---- workspace layout ----
    char* ws = (char*)d_ws;
    size_t off = 0;
    auto alloc = [&](size_t bytes) -> void* {
        void* p = ws + off;
        off = (off + bytes + 255) & ~(size_t)255;
        return p;
    };
    float*  wsAtt   = (float*) alloc((size_t)NTOK * DIM * sizeof(float));     // 256 MB
    half_t* wsXw    = (half_t*)alloc((size_t)NTOK * DIM * sizeof(half_t));    // 128 MB (reused as h2)
    half_t* wsQkvWt = (half_t*)alloc((size_t)768 * 256 * sizeof(half_t));
    half_t* wsProjWt= (half_t*)alloc((size_t)256 * 256 * sizeof(half_t));
    half_t* wsW1t   = (half_t*)alloc((size_t)1024 * 256 * sizeof(half_t));
    half_t* wsW2t   = (half_t*)alloc((size_t)256 * 1024 * sizeof(half_t));
    float*  wsBias  = (float*) alloc((size_t)HEADS * 64 * 64 * sizeof(float));

    // ---- raise dynamic LDS caps (host API, capture-safe) ----
    (void)hipFuncSetAttribute((const void*)k_window_attn,
        hipFuncAttributeMaxDynamicSharedMemorySize,
        LDS_ATTN_HALVES * (int)sizeof(half_t));
    (void)hipFuncSetAttribute((const void*)k_mlp_final,
        hipFuncAttributeMaxDynamicSharedMemorySize, (64 * DIM + 64 * MLP_HID) * 2);

    // ---- prep ----
    k_transpose_w<<<(768 * 256 + 255) / 256, 256, 0, stream>>>(qkv_w,  wsQkvWt, 256, 768);
    k_transpose_w<<<(256 * 256 + 255) / 256, 256, 0, stream>>>(proj_w, wsProjWt, 256, 256);
    k_transpose_w<<<(1024 * 256 + 255) / 256, 256, 0, stream>>>(mlp_w1, wsW1t, 256, 1024);
    k_transpose_w<<<(256 * 1024 + 255) / 256, 256, 0, stream>>>(mlp_w2, wsW2t, 1024, 256);
    k_build_bias<<<(HEADS * 64 * 64 + 255) / 256, 256, 0, stream>>>(rpb, wsBias);

    // ---- pipeline ----
    k_ln1_window<<<NWIN, 256, 0, stream>>>(x, ln1_g, ln1_b, wsXw);

    k_window_attn<<<NWIN, 256, LDS_ATTN_HALVES * sizeof(half_t), stream>>>(
        wsXw, wsQkvWt, qkv_b, wsProjWt, proj_b, wsBias, wsAtt);

    k_ln2<<<NTOK / 64, 256, 0, stream>>>(wsAtt, ln2_g, ln2_b, wsXw);

    k_mlp_final<<<NTOK / 64, 256, (64 * DIM + 64 * MLP_HID) * 2, stream>>>(
        wsXw, wsW1t, mlp_b1, wsW2t, mlp_b2, x, wsAtt, out);

    (void)n_in; (void)in_sizes; (void)out_size; (void)ws_size;
}
